// Block_11639361372686
// MI455X (gfx1250) — compile-verified, matrix operand order checked by reading, not run on hardware
//
#include <hip/hip_runtime.h>
#include <math.h>
#include <stdint.h>

// ---------------------------------------------------------------------------
// Types
// ---------------------------------------------------------------------------
typedef __bf16 bf16;
typedef __attribute__((ext_vector_type(16))) __bf16 v16bf;
typedef __attribute__((ext_vector_type(8)))  __bf16 v8bf;
typedef __attribute__((ext_vector_type(8)))  float  v8f;

#define DI static __device__ __forceinline__

// Model constants
constexpr int Bc = 2, Tc = 2048, Cc = 768, Hc = 12, HDc = 64;
constexpr int Mrows = Bc * Tc;          // 4096
constexpr int Nqkv  = 3 * Cc;           // 2304
constexpr int Ffc   = 4 * Cc;           // 3072

// ---------------------------------------------------------------------------
// WMMA / async helpers
// ---------------------------------------------------------------------------
DI v8f wmma_bf16(v16bf a, v16bf b, v8f c) {
  // D = A(16x32) * B(32x16) + C(16x16 f32)
  return __builtin_amdgcn_wmma_f32_16x16x32_bf16(
      /*neg_a=*/false, a, /*neg_b=*/false, b,
      /*c_mod=*/(short)0, c, /*reuse_a=*/false, /*reuse_b=*/false);
}

DI v8f vzero8() {
  v8f z;
#pragma unroll
  for (int i = 0; i < 8; ++i) z[i] = 0.0f;
  return z;
}

// Async global->LDS 16B copy (GLOBAL_LOAD_ASYNC_TO_LDS_B128, ASYNCcnt).
// LDS operand VGPR holds the wave-relative LDS byte address, which is the
// low 32 bits of the generic pointer to a __shared__ object.
DI void async_ld128(const void* g, void* lds) {
  unsigned laddr = (unsigned)(uintptr_t)lds;
  unsigned long long gaddr = (unsigned long long)(uintptr_t)g;
  asm volatile("global_load_async_to_lds_b128 %0, %1, off"
               :: "v"(laddr), "v"(gaddr)
               : "memory");
}
DI void wait_async0() { asm volatile("s_wait_asynccnt 0x0" ::: "memory"); }

// Load a 16x32 bf16 A/B^T fragment from an LDS tile whose rows are
// [row][k] with row-length >= k0+32.  Per ISA 7.12.2 (16-bit A 16x32):
// lanes 0-15 hold K=0..7 (v0..v3) and K=16..23 (v4..v7); lanes 16-31
// hold K=8..15 and K=24..31.  rowp = &tile[row_of_this_lane][0].
DI v16bf ld_frag(const bf16* rowp, int k0, int hl) {
  v8bf lo = *(const v8bf*)(rowp + k0 + hl * 8);
  v8bf hi = *(const v8bf*)(rowp + k0 + 16 + hl * 8);
  return __builtin_shufflevector(lo, hi,
      0, 1, 2, 3, 4, 5, 6, 7, 8, 9, 10, 11, 12, 13, 14, 15);
}

DI float redmax16(float v) {
#pragma unroll
  for (int m = 1; m < 16; m <<= 1) v = fmaxf(v, __shfl_xor(v, m, 32));
  return v;
}
DI float redsum16(float v) {
#pragma unroll
  for (int m = 1; m < 16; m <<= 1) v += __shfl_xor(v, m, 32);
  return v;
}

// ---------------------------------------------------------------------------
// Weight prep: f32 -> bf16, stored B-transposed  [N][K]
// ---------------------------------------------------------------------------
__global__ __launch_bounds__(256) void pack_qkv_w(
    const float* __restrict__ Wq, const float* __restrict__ Wk,
    const float* __restrict__ Wv, bf16* __restrict__ dst) {
  int idx = blockIdx.x * 256 + threadIdx.x;
  if (idx >= Nqkv * Cc) return;
  int n = idx / Cc, c = idx - n * Cc;        // dst[n][c], n in [0,2304)
  int which = n / Cc;                        // 0:q 1:k 2:v
  int r = n - which * Cc;                    // h*64+d
  int h = r >> 6, d = r & 63;
  const float* W = (which == 0) ? Wq : (which == 1) ? Wk : Wv;  // [H][C][HD]
  dst[idx] = (bf16)W[((size_t)h * Cc + c) * HDc + d];
}

__global__ __launch_bounds__(256) void pack_qkv_b(
    const float* __restrict__ bq, const float* __restrict__ bk,
    const float* __restrict__ bv, float* __restrict__ dst) {
  int idx = blockIdx.x * 256 + threadIdx.x;
  if (idx >= Nqkv) return;
  int which = idx / Cc, r = idx - which * Cc;
  const float* s = (which == 0) ? bq : (which == 1) ? bk : bv;
  dst[idx] = s[r];
}

// src [K][N] f32  ->  dst [N][K] bf16
__global__ __launch_bounds__(256) void transpose_cvt(
    const float* __restrict__ src, bf16* __restrict__ dst, int N, int K) {
  int idx = blockIdx.x * 256 + threadIdx.x;
  if (idx >= N * K) return;
  int n = idx / K, k = idx - n * K;
  dst[idx] = (bf16)src[(size_t)k * N + n];
}

// ---------------------------------------------------------------------------
// LayerNorm (one block per row of 768) -> bf16
// ---------------------------------------------------------------------------
__global__ __launch_bounds__(256) void ln_bf16_kernel(
    const float* __restrict__ x, const float* __restrict__ s,
    const float* __restrict__ b, bf16* __restrict__ out) {
  int row = blockIdx.x, tid = threadIdx.x;
  const float* xr = x + (size_t)row * Cc;
  float v0 = xr[tid], v1 = xr[tid + 256], v2 = xr[tid + 512];
  __shared__ float red[256];
  red[tid] = v0 + v1 + v2;
  __syncthreads();
#pragma unroll
  for (int off = 128; off > 0; off >>= 1) {
    if (tid < off) red[tid] += red[tid + off];
    __syncthreads();
  }
  float mu = red[0] * (1.0f / Cc);
  __syncthreads();
  float d0 = v0 - mu, d1 = v1 - mu, d2 = v2 - mu;
  red[tid] = d0 * d0 + d1 * d1 + d2 * d2;
  __syncthreads();
#pragma unroll
  for (int off = 128; off > 0; off >>= 1) {
    if (tid < off) red[tid] += red[tid + off];
    __syncthreads();
  }
  float rs = rsqrtf(red[0] * (1.0f / Cc) + 1e-5f);
  bf16* o = out + (size_t)row * Cc;
  o[tid]       = (bf16)(d0 * rs * s[tid]       + b[tid]);
  o[tid + 256] = (bf16)(d1 * rs * s[tid + 256] + b[tid + 256]);
  o[tid + 512] = (bf16)(d2 * rs * s[tid + 512] + b[tid + 512]);
}

// ---------------------------------------------------------------------------
// Tiled bf16 WMMA GEMM:  out[M][N] = A[M][K] * Bt[N][K]^T  (+ epilogue)
// Block = 256 threads (8 waves), tile 128x128, K-step 32, double-buffered
// LDS filled via GLOBAL_LOAD_ASYNC_TO_LDS_B128 (ASYNCcnt) so the next tile
// streams in while the matrix pipes run on the current tile.
// Wave (wm in 0..1, wn in 0..3) computes 64x32 via 4x2 WMMA frags.
// EPI: 0 = bias -> bf16 out; 1 = bias + residual -> f32 out;
//      2 = bias + exact GELU -> bf16 out
// ---------------------------------------------------------------------------
template <int EPI>
__global__ __launch_bounds__(256) void gemm_bf16_kernel(
    const bf16* __restrict__ A, const bf16* __restrict__ Bt,
    const float* __restrict__ bias, const float* __restrict__ res,
    void* __restrict__ outp, int M, int N, int K) {
  __shared__ bf16 As[2][128][32];
  __shared__ bf16 Bs[2][128][32];
  int tid = threadIdx.x, w = tid >> 5, ln = tid & 31;
  int hl = ln >> 4, l16 = ln & 15;
  int wm = w & 1, wn = w >> 1;
  int Mt = blockIdx.y * 128, Nt = blockIdx.x * 128;

  v8f acc[4][2];
#pragma unroll
  for (int i = 0; i < 4; ++i)
#pragma unroll
    for (int j = 0; j < 2; ++j) acc[i][j] = vzero8();

  int srow = tid >> 2, sseg = tid & 3;
  const bf16* Ag = A + (size_t)(Mt + srow) * K + sseg * 8;
  const bf16* Bg = Bt + (size_t)(Nt + srow) * K + sseg * 8;

  // Prologue: async-stage tile 0
  async_ld128(Ag, &As[0][srow][sseg * 8]);
  async_ld128(Bg, &Bs[0][srow][sseg * 8]);
  wait_async0();
  __syncthreads();

  int nk = K >> 5;
  for (int kt = 0; kt < nk; ++kt) {
    int buf = kt & 1;
    if (kt + 1 < nk) {  // async-stage next tile into the other buffer
      async_ld128(Ag + (kt + 1) * 32, &As[buf ^ 1][srow][sseg * 8]);
      async_ld128(Bg + (kt + 1) * 32, &Bs[buf ^ 1][srow][sseg * 8]);
    }
    if (kt + 8 < nk) {  // deep L2 prefetch (global_prefetch_b8)
      __builtin_prefetch(Ag + (kt + 8) * 32, 0, 0);
      __builtin_prefetch(Bg + (kt + 8) * 32, 0, 0);
    }

    v16bf af[4], bfm[2];
#pragma unroll
    for (int ms = 0; ms < 4; ++ms)
      af[ms] = ld_frag(&As[buf][wm * 64 + ms * 16 + l16][0], 0, hl);
#pragma unroll
    for (int ns = 0; ns < 2; ++ns)
      bfm[ns] = ld_frag(&Bs[buf][wn * 32 + ns * 16 + l16][0], 0, hl);
#pragma unroll
    for (int ms = 0; ms < 4; ++ms)
#pragma unroll
      for (int ns = 0; ns < 2; ++ns)
        acc[ms][ns] = wmma_bf16(af[ms], bfm[ns], acc[ms][ns]);

    wait_async0();    // next tile landed in LDS
    __syncthreads();  // all waves done reading current tile
  }

  // Epilogue. C/D layout: VGPR r holds M=r (lanes 0-15) / r+8 (lanes 16-31),
  // N = lane%16.
#pragma unroll
  for (int ms = 0; ms < 4; ++ms)
#pragma unroll
    for (int ns = 0; ns < 2; ++ns)
#pragma unroll
      for (int r = 0; r < 8; ++r) {
        int grow = Mt + wm * 64 + ms * 16 + r + 8 * hl;
        int gcol = Nt + wn * 32 + ns * 16 + l16;
        float v = acc[ms][ns][r] + bias[gcol];
        size_t oidx = (size_t)grow * N + gcol;
        if (EPI == 1) {
          ((float*)outp)[oidx] = v + res[oidx];
        } else if (EPI == 2) {
          float g = 0.5f * v * (1.0f + erff(v * 0.70710678118654752f));
          ((bf16*)outp)[oidx] = (bf16)g;
        } else {
          ((bf16*)outp)[oidx] = (bf16)v;
        }
      }
}

// ---------------------------------------------------------------------------
// Flash attention: block = (b, h, 128 query rows), 8 waves x 16 rows.
// qkv buffer: [4096][2304] bf16; q at col h*64, k at 768+h*64, v at 1536+h*64.
// Key blocks of 64; online softmax; P staged per-wave in LDS to re-enter
// WMMA in A-fragment layout.  Q/K tiles staged via async-to-LDS (ASYNCcnt),
// V transpose staging via ds stores (DScnt).  y out: [B,T,C] bf16.
// ---------------------------------------------------------------------------
__global__ __launch_bounds__(256) void attn_kernel(
    const bf16* __restrict__ qkv, const unsigned char* __restrict__ maskp,
    bf16* __restrict__ y) {
  __shared__ bf16 Qs[128][64];     // 16 KB
  __shared__ bf16 Ks[64][64];      //  8 KB  (K rows [s][d] == B^T for Q.K^T)
  __shared__ bf16 Vt[64][64];      //  8 KB  (V^T [d][s] == B^T for P.V)
  __shared__ bf16 Ps[8][16][64];   // 16 KB  (per-wave P tile)

  int tid = threadIdx.x, w = tid >> 5, ln = tid & 31;
  int hl = ln >> 4, l16 = ln & 15;
  int qt = blockIdx.x, h = blockIdx.y, b = blockIdx.z;
  int q0 = qt * 128;
  size_t rowbase = (size_t)b * Tc;

  // Stage Q tile (128 x 64 bf16) via async copies; completion is covered by
  // the wait_async0 + barrier at the top of the first key-block iteration.
#pragma unroll
  for (int i = 0; i < 4; ++i) {
    int chunk = tid + i * 256;          // 0..1023
    int row = chunk >> 3, seg = chunk & 7;
    async_ld128(qkv + (rowbase + q0 + row) * Nqkv + h * 64 + seg * 8,
                &Qs[row][seg * 8]);
  }

  float mrow[8], lrow[8];
  v8f ov[4];
#pragma unroll
  for (int r = 0; r < 8; ++r) { mrow[r] = -1e30f; lrow[r] = 0.0f; }
#pragma unroll
  for (int nd = 0; nd < 4; ++nd) ov[nd] = vzero8();

  for (int s0 = 0; s0 < Tc; s0 += 64) {
    __syncthreads();  // all waves done reading Ks/Vt from previous iteration
    // Stage K tile rows [s][d] (async -> LDS)
#pragma unroll
    for (int i = 0; i < 2; ++i) {
      int chunk = tid + i * 256;        // 0..511
      int row = chunk >> 3, seg = chunk & 7;
      async_ld128(qkv + (rowbase + s0 + row) * Nqkv + Cc + h * 64 + seg * 8,
                  &Ks[row][seg * 8]);
    }
    // Stage V transposed: Vt[d][s] (vmem loads + ds stores)
#pragma unroll
    for (int i = 0; i < 16; ++i) {
      int idx = tid * 16 + i;           // 0..4095
      int s = idx >> 6, d = idx & 63;
      Vt[d][s] = qkv[(rowbase + s0 + s) * Nqkv + 2 * Cc + h * 64 + d];
    }
    float addm[4];
#pragma unroll
    for (int ns = 0; ns < 4; ++ns) {
      int s = s0 + ns * 16 + l16;
      addm[ns] = maskp[(size_t)b * Tc + s] ? -1e30f : 0.0f;
    }
    wait_async0();
    __syncthreads();

    // S = Q * K^T  (wave's 16 query rows x 64 keys)
    v8f sc[4];
#pragma unroll
    for (int ns = 0; ns < 4; ++ns) sc[ns] = vzero8();
#pragma unroll
    for (int kb = 0; kb < 2; ++kb) {
      v16bf aq = ld_frag(&Qs[w * 16 + l16][0], kb * 32, hl);
#pragma unroll
      for (int ns = 0; ns < 4; ++ns) {
        v16bf bk = ld_frag(&Ks[ns * 16 + l16][0], kb * 32, hl);
        sc[ns] = wmma_bf16(aq, bk, sc[ns]);
      }
    }

    // Online softmax update (rows r+8*hl, reductions within 16-lane halves)
    float esc[8];
#pragma unroll
    for (int r = 0; r < 8; ++r) {
      float mx = mrow[r];
#pragma unroll
      for (int ns = 0; ns < 4; ++ns)
        mx = fmaxf(mx, sc[ns][r] * 0.125f + addm[ns]);
      mx = redmax16(mx);
      esc[r] = __expf(mrow[r] - mx);
      mrow[r] = mx;
    }
    float psum[8];
#pragma unroll
    for (int r = 0; r < 8; ++r) psum[r] = 0.0f;
#pragma unroll
    for (int ns = 0; ns < 4; ++ns)
#pragma unroll
      for (int r = 0; r < 8; ++r) {
        float pv = __expf(sc[ns][r] * 0.125f + addm[ns] - mrow[r]);
        psum[r] += pv;
        Ps[w][r + 8 * hl][ns * 16 + l16] = (bf16)pv;
      }
#pragma unroll
    for (int r = 0; r < 8; ++r)
      lrow[r] = lrow[r] * esc[r] + redsum16(psum[r]);
#pragma unroll
    for (int nd = 0; nd < 4; ++nd)
#pragma unroll
      for (int r = 0; r < 8; ++r) ov[nd][r] *= esc[r];

    // O += P * V   (K-dim = 64 keys, N-dim = 64 dims)
#pragma unroll
    for (int kb = 0; kb < 2; ++kb) {
      v16bf ap = ld_frag(&Ps[w][l16][0], kb * 32, hl);
#pragma unroll
      for (int nd = 0; nd < 4; ++nd) {
        v16bf bv = ld_frag(&Vt[nd * 16 + l16][0], kb * 32, hl);
        ov[nd] = wmma_bf16(ap, bv, ov[nd]);
      }
    }
  }

  // Normalize and store to y[B,T,C] at column h*64+d
#pragma unroll
  for (int nd = 0; nd < 4; ++nd)
#pragma unroll
    for (int r = 0; r < 8; ++r) {
      int t = q0 + w * 16 + r + 8 * hl;
      int d = nd * 16 + l16;
      y[(rowbase + t) * Cc + h * 64 + d] = (bf16)(ov[nd][r] / lrow[r]);
    }
}

// ---------------------------------------------------------------------------
// Host launcher
// ---------------------------------------------------------------------------
extern "C" void kernel_launch(void* const* d_in, const int* in_sizes, int n_in,
                              void* d_out, int out_size, void* d_ws,
                              size_t ws_size, hipStream_t stream) {
  (void)in_sizes; (void)n_in; (void)out_size; (void)ws_size;
  const float*         x     = (const float*)d_in[0];
  const unsigned char* mask  = (const unsigned char*)d_in[1];
  const float* ln1s  = (const float*)d_in[2];
  const float* ln1b  = (const float*)d_in[3];
  const float* Wq    = (const float*)d_in[4];
  const float* bq    = (const float*)d_in[5];
  const float* Wk    = (const float*)d_in[6];
  const float* bk    = (const float*)d_in[7];
  const float* Wv    = (const float*)d_in[8];
  const float* bv    = (const float*)d_in[9];
  const float* Wo    = (const float*)d_in[10];
  const float* bo    = (const float*)d_in[11];
  const float* ln2s  = (const float*)d_in[12];
  const float* ln2b  = (const float*)d_in[13];
  const float* Wfc   = (const float*)d_in[14];
  const float* bfc   = (const float*)d_in[15];
  const float* Wproj = (const float*)d_in[16];
  const float* bproj = (const float*)d_in[17];

  char* base = (char*)d_ws;
  size_t off = 0;
  auto alloc = [&](size_t bytes) -> void* {
    void* p = base + off;
    off += (bytes + 255) & ~(size_t)255;
    return p;
  };
  bf16*  wqkvT  = (bf16*)alloc((size_t)Nqkv * Cc * 2);
  float* bqkv   = (float*)alloc((size_t)Nqkv * 4);
  bf16*  woT    = (bf16*)alloc((size_t)Cc * Cc * 2);
  bf16*  wfcT   = (bf16*)alloc((size_t)Ffc * Cc * 2);
  bf16*  wprojT = (bf16*)alloc((size_t)Cc * Ffc * 2);
  bf16*  xn     = (bf16*)alloc((size_t)Mrows * Cc * 2);   // reused for xn2
  bf16*  big    = (bf16*)alloc((size_t)Mrows * Ffc * 2);  // qkv, then h
  bf16*  ybuf   = (bf16*)alloc((size_t)Mrows * Cc * 2);
  float* x1     = (float*)alloc((size_t)Mrows * Cc * 4);

  // --- weight prep ---
  pack_qkv_w<<<(Nqkv * Cc + 255) / 256, 256, 0, stream>>>(Wq, Wk, Wv, wqkvT);
  pack_qkv_b<<<(Nqkv + 255) / 256, 256, 0, stream>>>(bq, bk, bv, bqkv);
  transpose_cvt<<<(Cc * Cc + 255) / 256, 256, 0, stream>>>(Wo, woT, Cc, Cc);
  transpose_cvt<<<(Ffc * Cc + 255) / 256, 256, 0, stream>>>(Wfc, wfcT, Ffc, Cc);
  transpose_cvt<<<(Cc * Ffc + 255) / 256, 256, 0, stream>>>(Wproj, wprojT, Cc, Ffc);

  // --- attention sublayer ---
  ln_bf16_kernel<<<Mrows, 256, 0, stream>>>(x, ln1s, ln1b, xn);
  gemm_bf16_kernel<0><<<dim3(Nqkv / 128, Mrows / 128), 256, 0, stream>>>(
      xn, wqkvT, bqkv, nullptr, (void*)big, Mrows, Nqkv, Cc);
  attn_kernel<<<dim3(Tc / 128, Hc, Bc), 256, 0, stream>>>(big, mask, ybuf);
  gemm_bf16_kernel<1><<<dim3(Cc / 128, Mrows / 128), 256, 0, stream>>>(
      ybuf, woT, bo, x, (void*)x1, Mrows, Cc, Cc);

  // --- MLP sublayer ---
  ln_bf16_kernel<<<Mrows, 256, 0, stream>>>(x1, ln2s, ln2b, xn);
  gemm_bf16_kernel<2><<<dim3(Ffc / 128, Mrows / 128), 256, 0, stream>>>(
      xn, wfcT, bfc, nullptr, (void*)big, Mrows, Ffc, Cc);
  gemm_bf16_kernel<1><<<dim3(Cc / 128, Mrows / 128), 256, 0, stream>>>(
      big, wprojT, bproj, x1, d_out, Mrows, Cc, Ffc);
}
